// DeltaNet_31877247271476
// MI455X (gfx1250) — compile-verified, matrix operand order checked by reading, not run on hardware
//
#include <hip/hip_runtime.h>

#define B_  4
#define L_  4096
#define D_  1024
#define H_  4
#define DK_ 256
#define DV_ 256
#define HID_ 512
#define CHUNK_ 32
#define DVL_ 32        // dv slice per delta workgroup
#define SPLIT_ 8       // DV_/DVL_
#define LDST_ (DK_ + 8)  // padded row pitch of transposed state

typedef _Float16 v16h __attribute__((ext_vector_type(16)));
typedef _Float16 v8h  __attribute__((ext_vector_type(8)));
typedef float    v8f  __attribute__((ext_vector_type(8)));
typedef int      v4i_ __attribute__((ext_vector_type(4)));

__device__ __forceinline__ v8f wmma16(v16h a, v16h b, v8f c) {
  return __builtin_amdgcn_wmma_f32_16x16x32_f16(false, a, false, b, (short)0, c, false, false);
}

// ---------------------------------------------------------------------------
// Async global->LDS staging (CDNA5 GLOBAL_LOAD_ASYNC_TO_LDS_B128, ASYNCcnt).
// Builtin signature (from compiler diagnostic): (int4 AS1*, int4 AS3*, imm, imm)
// ---------------------------------------------------------------------------
#if __has_builtin(__builtin_amdgcn_global_load_async_to_lds_b128)
#define HAS_ASYNC_LDS 1
#else
#define HAS_ASYNC_LDS 0
#endif

__device__ __forceinline__ void async_cp16(const float* g, float* l) {
#if HAS_ASYNC_LDS
  __builtin_amdgcn_global_load_async_to_lds_b128(
      (__attribute__((address_space(1))) v4i_*)g,
      (__attribute__((address_space(3))) v4i_*)l, 0, 0);
#else
  float4 t = *(const float4*)g;
  *(float4*)l = t;
#endif
}

template <int N>
__device__ __forceinline__ void async_wait_n() {
#if HAS_ASYNC_LDS
#if __has_builtin(__builtin_amdgcn_s_wait_asynccnt)
  __builtin_amdgcn_s_wait_asynccnt(N);
#else
  asm volatile("s_wait_asynccnt %0" ::"i"(N) : "memory");
#endif
#endif
}

// ---------------------------------------------------------------------------
// Fragment loaders. CDNA5 16-bit A/B layout: lane&15 -> row; each lane holds
// two contiguous 8xf16 runs: K = kbase+{0..7} and kbase+{16..23}, where
// kbase = (lane<16) ? 0 : 8. So a fragment is two 16-byte LDS loads.
// ---------------------------------------------------------------------------
__device__ __forceinline__ v16h frag_mk16(const _Float16* base, int ld, int r0,
                                          int k0, int lane) {
  const _Float16* p =
      base + (size_t)(r0 + (lane & 15)) * ld + k0 + ((lane & 16) ? 8 : 0);
  v8h lo = *(const v8h*)(p);
  v8h hi = *(const v8h*)(p + 16);
  return __builtin_shufflevector(lo, hi, 0, 1, 2, 3, 4, 5, 6, 7, 8, 9, 10, 11,
                                 12, 13, 14, 15);
}

// Same pattern from an fp32 row-major array, converting to f16 on load.
__device__ __forceinline__ v16h frag_mk32(const float* base, int ld, int r0,
                                          int k0, int lane) {
  const float* p =
      base + (size_t)(r0 + (lane & 15)) * ld + k0 + ((lane & 16) ? 8 : 0);
  v8f lo = *(const v8f*)(p);
  v8f hi = *(const v8f*)(p + 16);
  return __builtin_shufflevector(__builtin_convertvector(lo, v8h),
                                 __builtin_convertvector(hi, v8h), 0, 1, 2, 3,
                                 4, 5, 6, 7, 8, 9, 10, 11, 12, 13, 14, 15);
}

__device__ __forceinline__ float sigmoidf_(float x) { return 1.f / (1.f + __expf(-x)); }
__device__ __forceinline__ float siluf_(float x)    { return x / (1.f + __expf(-x)); }

// ---------------------------------------------------------------------------
// GEMM: Y[M,N] = act(X[M,K] @ W[N,K]^T + bias), fp32 in/out, f16 WMMA compute.
// 256 threads (8 waves), 128x128 tile, K-step 32, double-buffered async stage.
// Each wave owns a 32x64 output block -> 8 WMMA per K-step.
// ---------------------------------------------------------------------------
template <int ACT>
__global__ __launch_bounds__(256) void gemm_xwt(const float* __restrict__ X,
                                                const float* __restrict__ W,
                                                const float* __restrict__ bias,
                                                float* __restrict__ Y,
                                                int M, int N, int K) {
  __shared__ alignas(32) float sX[2][128][32];
  __shared__ alignas(32) float sW[2][128][32];
  const int tid = threadIdx.x, lane = tid & 31, wv = tid >> 5;
  const int m0 = (int)blockIdx.y * 128;
  const int n0 = (int)blockIdx.x * 128;
  const int rowA = (wv >> 1) * 32;     // wave's 32 rows
  const int colB = (wv & 1) * 64;      // wave's 64 cols
  v8f acc[2][4];
#pragma unroll
  for (int i = 0; i < 2; ++i)
#pragma unroll
    for (int j = 0; j < 4; ++j) acc[i][j] = {};

  auto stage = [&](int kt, int b) {
    const int k0 = kt * 32;
#pragma unroll
    for (int i = 0; i < 4; ++i) {
      int li = tid + i * 256;           // 1024 float4 per tile
      int rr = li >> 3, c4 = (li & 7) * 4;
      async_cp16(X + (size_t)(m0 + rr) * K + k0 + c4, &sX[b][rr][c4]);
      async_cp16(W + (size_t)(n0 + rr) * K + k0 + c4, &sW[b][rr][c4]);
    }
  };

  const int NT = K / 32;
  stage(0, 0);
  for (int kt = 0; kt < NT; ++kt) {
    const int cur = kt & 1;
    if (kt + 1 < NT) {
      stage(kt + 1, cur ^ 1);
      async_wait_n<8>();     // the 8 copies of tile kt are complete (in-order)
    } else {
      async_wait_n<0>();
    }
    __syncthreads();
    v16h a0 = frag_mk32(&sX[cur][0][0], 32, rowA, 0, lane);
    v16h a1 = frag_mk32(&sX[cur][0][0], 32, rowA + 16, 0, lane);
#pragma unroll
    for (int j = 0; j < 4; ++j) {
      v16h bj = frag_mk32(&sW[cur][0][0], 32, colB + 16 * j, 0, lane);
      acc[0][j] = wmma16(a0, bj, acc[0][j]);
      acc[1][j] = wmma16(a1, bj, acc[1][j]);
    }
    __syncthreads();
  }
  const int hi = (lane >> 4) & 1;
#pragma unroll
  for (int i = 0; i < 2; ++i)
#pragma unroll
    for (int j = 0; j < 4; ++j) {
      const int n = n0 + colB + 16 * j + (lane & 15);
      const float bv = bias ? bias[n] : 0.f;
#pragma unroll
      for (int r = 0; r < 8; ++r) {
        int m = m0 + rowA + 16 * i + r + 8 * hi;
        float v = acc[i][j][r] + bv;
        if (ACT == 1) v = siluf_(v);
        Y[(size_t)m * N + n] = v;
      }
    }
}

// ---------------------------------------------------------------------------
// Causal depthwise conv (k=4) + SiLU, layout (B,L,D).
// ---------------------------------------------------------------------------
__global__ void conv_silu_kernel(const float* __restrict__ x,
                                 const float* __restrict__ w,
                                 float* __restrict__ y, int total) {
  for (int idx = blockIdx.x * blockDim.x + threadIdx.x; idx < total;
       idx += gridDim.x * blockDim.x) {
    int d = idx % D_;
    int bl = idx / D_;
    int l = bl % L_;
    float acc = 0.f;
#pragma unroll
    for (int j = 0; j < 4; ++j) {
      int ls = l - 3 + j;
      if (ls >= 0) acc += w[d * 4 + j] * x[(size_t)(bl - 3 + j) * D_ + d];
    }
    y[idx] = siluf_(acc);
  }
}

// ---------------------------------------------------------------------------
// beta / g_s / g_l: one wave per (b,l) row, 12 dots of length D.
// ---------------------------------------------------------------------------
__global__ __launch_bounds__(256) void smallproj_kernel(
    const float* __restrict__ x, const float* __restrict__ Wb,
    const float* __restrict__ Wds, const float* __restrict__ bds,
    const float* __restrict__ Wdl, const float* __restrict__ bdl,
    float* __restrict__ beta, float* __restrict__ gs, float* __restrict__ gl) {
  const int row = blockIdx.x * 8 + (threadIdx.x >> 5);
  const int lane = threadIdx.x & 31;
  const float* xr = x + (size_t)row * D_;
  float acc[12];
#pragma unroll
  for (int i = 0; i < 12; ++i) acc[i] = 0.f;
  for (int kk = lane; kk < D_; kk += 32) {
    float xv = xr[kk];
#pragma unroll
    for (int h = 0; h < H_; ++h) {
      acc[h]     += xv * Wb [h * D_ + kk];
      acc[4 + h] += xv * Wds[h * D_ + kk];
      acc[8 + h] += xv * Wdl[h * D_ + kk];
    }
  }
#pragma unroll
  for (int i = 0; i < 12; ++i)
    for (int off = 16; off; off >>= 1) acc[i] += __shfl_xor(acc[i], off, 32);
  if (lane == 0) {
#pragma unroll
    for (int h = 0; h < H_; ++h) {
      beta[(size_t)row * H_ + h] = sigmoidf_(acc[h]);
      gs  [(size_t)row * H_ + h] = sigmoidf_(acc[4 + h] + bds[h]);
      gl  [(size_t)row * H_ + h] = sigmoidf_(acc[8 + h] + bdl[h]);
    }
  }
}

// ---------------------------------------------------------------------------
// Hierarchical gate: one wave per (b,l) row, 24 dots of length HID.
// ---------------------------------------------------------------------------
__global__ __launch_bounds__(256) void gate_kernel(
    const float* __restrict__ z, const float* __restrict__ Wc,
    const float* __restrict__ bc, const float* __restrict__ Wl,
    const float* __restrict__ bl, const float* __restrict__ Wg,
    const float* __restrict__ bg, const float* __restrict__ ltc,
    const float* __restrict__ ltf, float* __restrict__ wout) {
  const int row = blockIdx.x * 8 + (threadIdx.x >> 5);
  const int lane = threadIdx.x & 31;
  const float* zr = z + (size_t)row * HID_;
  float acc[24];
#pragma unroll
  for (int i = 0; i < 24; ++i) acc[i] = 0.f;
  for (int kk = lane; kk < HID_; kk += 32) {
    float zv = zr[kk];
#pragma unroll
    for (int o = 0; o < 8; ++o) {
      acc[o]      += zv * Wc[o * HID_ + kk];
      acc[8 + o]  += zv * Wl[o * HID_ + kk];
      acc[16 + o] += zv * Wg[o * HID_ + kk];
    }
  }
#pragma unroll
  for (int i = 0; i < 24; ++i)
    for (int off = 16; off; off >>= 1) acc[i] += __shfl_xor(acc[i], off, 32);
  if (lane == 0) {
#pragma unroll
    for (int h = 0; h < H_; ++h) {
      float tc = log1pf(__expf(ltc[h])) + 1e-4f;
      float tf = log1pf(__expf(ltf[h])) + 1e-4f;
      float a0 = (acc[2 * h] + bc[2 * h]) / tc;
      float a1 = (acc[2 * h + 1] + bc[2 * h + 1]) / tc;
      float mm = fmaxf(a0, a1);
      float e0 = __expf(a0 - mm), e1 = __expf(a1 - mm);
      float pc0 = e0 / (e0 + e1), pc1 = e1 / (e0 + e1);
      float q0a = (acc[8 + 2 * h] + bl[2 * h]) / tf;
      float q1a = (acc[8 + 2 * h + 1] + bl[2 * h + 1]) / tf;
      mm = fmaxf(q0a, q1a); e0 = __expf(q0a - mm); e1 = __expf(q1a - mm);
      float qq0 = e0 / (e0 + e1), qq1 = e1 / (e0 + e1);
      float r0a = (acc[16 + 2 * h] + bg[2 * h]) / tf;
      float r1a = (acc[16 + 2 * h + 1] + bg[2 * h + 1]) / tf;
      mm = fmaxf(r0a, r1a); e0 = __expf(r0a - mm); e1 = __expf(r1a - mm);
      float rr0 = e0 / (e0 + e1), rr1 = e1 / (e0 + e1);
      size_t o = ((size_t)row * H_ + h) * 4;
      wout[o + 0] = pc0 * qq0;   // V
      wout[o + 1] = pc0 * qq1;   // Es
      wout[o + 2] = pc1 * rr0;   // D
      wout[o + 3] = pc1 * rr1;   // El
    }
  }
}

// ---------------------------------------------------------------------------
// Chunkwise delta rule. One workgroup per (b,h,dv-slice). State kept
// TRANSPOSED in LDS: sST[dv][dk] fp32 with padded pitch, so S B-fragments are
// vector loads and the S update is a contiguous v8f RMW. q/k staged via async
// copies; K kept both row-major (sK) and transposed (sKT) for the kT matmul.
// ---------------------------------------------------------------------------
__global__ __launch_bounds__(256) void delta_kernel(
    const float* __restrict__ qg, const float* __restrict__ kg,
    const float* __restrict__ vg, const float* __restrict__ betag,
    float* __restrict__ og) {
  __shared__ alignas(32) float    sST[DVL_][LDST_];   // ~33 KB fp32 state (dv, dk)
  __shared__ alignas(32) float    sStage[CHUNK_][DK_]; // 32 KB async staging
  __shared__ alignas(32) _Float16 sQ[CHUNK_][DK_];    // normalized q
  __shared__ alignas(32) _Float16 sK[CHUNK_][DK_];    // normalized k
  __shared__ alignas(32) _Float16 sKT[DK_][CHUNK_];   // k transposed
  __shared__ alignas(32) _Float16 sVT[DVL_][CHUNK_];  // (beta*v)^T
  __shared__ alignas(32) _Float16 sUT[DVL_][CHUNK_];  // u^T
  __shared__ alignas(32) _Float16 sUhT[DVL_][CHUNK_]; // uhat^T
  __shared__ alignas(32) _Float16 sPT[DVL_][CHUNK_];  // ((beta*k)@S)^T
  __shared__ alignas(32) _Float16 sAttn[CHUNK_][CHUNK_];
  __shared__ alignas(32) _Float16 sTinv[CHUNK_][CHUNK_];
  __shared__ float sLm[CHUNK_][CHUNK_];
  __shared__ float sBeta[CHUNK_];
  __shared__ float sRed[CHUNK_][8];
  __shared__ float sNrm[2][CHUNK_];

  const int tid = threadIdx.x, lane = tid & 31, wv = tid >> 5;
  const int bh = blockIdx.x / SPLIT_, sp = blockIdx.x % SPLIT_;
  const int bb = bh / H_, hh = bh % H_;
  const int dv0 = sp * DVL_;
  const size_t baseQK = (size_t)bb * L_ * D_ + (size_t)hh * DK_;
  const size_t baseV  = (size_t)bb * L_ * D_ + (size_t)hh * DV_ + dv0;

  for (int i = tid; i < DVL_ * LDST_; i += 256) (&sST[0][0])[i] = 0.f;
  __syncthreads();

  for (int ch = 0; ch < L_ / CHUNK_; ++ch) {
    const int l0 = ch * CHUNK_;
    const int r = tid >> 3, p8 = tid & 7;   // 8 threads per chunk row

    // ---------- phase 1a: stage q (async), l2norm, f16 store ----------
    if (tid < CHUNK_) sBeta[tid] = betag[((size_t)bb * L_ + l0 + tid) * H_ + hh];
#pragma unroll
    for (int i = 0; i < 8; ++i) {
      int li = tid + i * 256;               // 2048 float4 slots
      int rr = li >> 6, c4 = (li & 63) * 4;
      async_cp16(qg + baseQK + (size_t)(l0 + rr) * D_ + c4, &sStage[rr][c4]);
    }
    async_wait_n<0>();
    __syncthreads();
    {
      float aq = 0.f;
      for (int j = 0; j < 32; ++j) { float x = sStage[r][p8 * 32 + j]; aq += x * x; }
      sRed[r][p8] = aq;
    }
    __syncthreads();
    if (tid < CHUNK_) {
      float s = 0.f;
      for (int j = 0; j < 8; ++j) s += sRed[tid][j];
      sNrm[0][tid] = rsqrtf(s + 1e-6f);
    }
    __syncthreads();
    {
      const float qn = sNrm[0][r];
      for (int j = 0; j < 32; ++j) {
        int c = p8 * 32 + j;
        sQ[r][c] = (_Float16)(sStage[r][c] * qn);
      }
      for (int i = tid; i < CHUNK_ * DVL_; i += 256) {
        int rr = i >> 5, cc = i & 31;
        sVT[cc][rr] = (_Float16)(vg[baseV + (size_t)(l0 + rr) * D_ + cc] * sBeta[rr]);
      }
    }
    __syncthreads();

    // ---------- phase 1b: stage k (async), l2norm, f16 store (+transpose) ----
#pragma unroll
    for (int i = 0; i < 8; ++i) {
      int li = tid + i * 256;
      int rr = li >> 6, c4 = (li & 63) * 4;
      async_cp16(kg + baseQK + (size_t)(l0 + rr) * D_ + c4, &sStage[rr][c4]);
    }
    async_wait_n<0>();
    __syncthreads();
    {
      float ak = 0.f;
      for (int j = 0; j < 32; ++j) { float x = sStage[r][p8 * 32 + j]; ak += x * x; }
      sRed[r][p8] = ak;
    }
    __syncthreads();
    if (tid < CHUNK_) {
      float s = 0.f;
      for (int j = 0; j < 8; ++j) s += sRed[tid][j];
      sNrm[1][tid] = rsqrtf(s + 1e-6f);
    }
    __syncthreads();
    {
      const float kn = sNrm[1][r];
      for (int j = 0; j < 32; ++j) {
        int c = p8 * 32 + j;
        _Float16 hv = (_Float16)(sStage[r][c] * kn);
        sK[r][c] = hv;
        sKT[c][r] = hv;
      }
    }
    __syncthreads();

    // ---------- phase 2: Lm = tril(bk k^T,-1) [w0-3]; attn = mask(q k^T) [w4-7] ----
    {
      const int w2 = (wv & 3);
      const int r0 = (w2 >> 1) * 16, n0 = (w2 & 1) * 16;
      v8f acc = {};
      if (wv < 4) {
        const _Float16 bsc = (_Float16)sBeta[r0 + (lane & 15)];
        for (int k0 = 0; k0 < DK_; k0 += 32) {
          v16h a = frag_mk16(&sK[0][0], DK_, r0, k0, lane);
#pragma unroll
          for (int e = 0; e < 16; ++e) a[e] = a[e] * bsc;
          v16h b = frag_mk16(&sK[0][0], DK_, n0, k0, lane);
          acc = wmma16(a, b, acc);
        }
        const int n = n0 + (lane & 15), hi = (lane >> 4) & 1;
#pragma unroll
        for (int rr = 0; rr < 8; ++rr) {
          int m = r0 + rr + 8 * hi;
          sLm[m][n] = (n < m) ? acc[rr] : 0.f;
        }
      } else {
        for (int k0 = 0; k0 < DK_; k0 += 32) {
          v16h a = frag_mk16(&sQ[0][0], DK_, r0, k0, lane);
          v16h b = frag_mk16(&sK[0][0], DK_, n0, k0, lane);
          acc = wmma16(a, b, acc);
        }
        const int n = n0 + (lane & 15), hi = (lane >> 4) & 1;
#pragma unroll
        for (int rr = 0; rr < 8; ++rr) {
          int m = r0 + rr + 8 * hi;
          sAttn[m][n] = (n <= m) ? (_Float16)acc[rr] : (_Float16)0.f;
        }
      }
    }
    __syncthreads();

    // ---------- phase 3: Tinv (wave0, fp32) || P^T = ((bk)@S)^T [w4-7] ----------
    if (wv == 0) {
      float t[32];
#pragma unroll
      for (int i = 0; i < 32; ++i) t[i] = (i == lane) ? 1.f : 0.f;
      for (int i = 1; i < 32; ++i) {
        float s = 0.f;
        for (int j = 0; j < i; ++j) s += sLm[i][j] * t[j];
        t[i] -= s;
      }
#pragma unroll
      for (int i = 0; i < 32; ++i) sTinv[i][lane] = (_Float16)t[i];
    } else if (wv >= 4) {
      const int w2 = wv - 4;
      const int r0 = (w2 >> 1) * 16, n0 = (w2 & 1) * 16;
      const _Float16 bsc = (_Float16)sBeta[r0 + (lane & 15)];
      v8f acc = {};
      for (int k0 = 0; k0 < DK_; k0 += 32) {
        v16h a = frag_mk16(&sK[0][0], DK_, r0, k0, lane);
#pragma unroll
        for (int e = 0; e < 16; ++e) a[e] = a[e] * bsc;
        v16h b = frag_mk32(&sST[0][0], LDST_, n0, k0, lane);  // B[kk][n] = sST[n][kk]
        acc = wmma16(a, b, acc);
      }
      const int n = n0 + (lane & 15), hi = (lane >> 4) & 1;
      *(v8h*)&sPT[n][r0 + 8 * hi] = __builtin_convertvector(acc, v8h);
    }
    __syncthreads();

    // ---------- phase 4: u = Tinv@v [w0-3]; wS = Tinv@P [w4-7, keep acc] ----------
    v8f accW = {};
    {
      const int w2 = (wv & 3);
      const int r0 = (w2 >> 1) * 16, n0 = (w2 & 1) * 16;
      if (wv < 4) {
        v16h a = frag_mk16(&sTinv[0][0], CHUNK_, r0, 0, lane);
        v16h b = frag_mk16(&sVT[0][0], CHUNK_, n0, 0, lane);
        v8f acc = {};
        acc = wmma16(a, b, acc);
        const int n = n0 + (lane & 15), hi = (lane >> 4) & 1;
        *(v8h*)&sUT[n][r0 + 8 * hi] = __builtin_convertvector(acc, v8h);
      } else {
        v16h a = frag_mk16(&sTinv[0][0], CHUNK_, r0, 0, lane);
        v16h b = frag_mk16(&sPT[0][0], CHUNK_, n0, 0, lane);
        accW = wmma16(a, b, accW);
      }
    }
    __syncthreads();

    // ---------- phase 5: uhat^T = (u - wS)^T [w4-7] ----------
    if (wv >= 4) {
      const int w2 = wv - 4;
      const int r0 = (w2 >> 1) * 16, n0 = (w2 & 1) * 16;
      const int n = n0 + (lane & 15), hi = (lane >> 4) & 1;
      v8h uv = *(const v8h*)&sUT[n][r0 + 8 * hi];
      v8f uh;
#pragma unroll
      for (int rr = 0; rr < 8; ++rr) uh[rr] = (float)uv[rr] - accW[rr];
      *(v8h*)&sUhT[n][r0 + 8 * hi] = __builtin_convertvector(uh, v8h);
    }
    __syncthreads();

    // ---------- phase 6: o = q@S + attn@uhat [w0-3] (reads OLD S) ----------
    if (wv < 4) {
      const int r0 = ((wv & 3) >> 1) * 16, n0 = (wv & 1) * 16;
      v8f acc = {};
      for (int k0 = 0; k0 < DK_; k0 += 32) {
        v16h a = frag_mk16(&sQ[0][0], DK_, r0, k0, lane);
        v16h b = frag_mk32(&sST[0][0], LDST_, n0, k0, lane);
        acc = wmma16(a, b, acc);
      }
      {
        v16h a = frag_mk16(&sAttn[0][0], CHUNK_, r0, 0, lane);
        v16h b = frag_mk16(&sUhT[0][0], CHUNK_, n0, 0, lane);
        acc = wmma16(a, b, acc);
      }
      const int n = n0 + (lane & 15), hi = (lane >> 4) & 1;
#pragma unroll
      for (int rr = 0; rr < 8; ++rr) {
        int m = r0 + rr + 8 * hi;
        og[baseV + (size_t)(l0 + m) * D_ + n] = acc[rr];
      }
    }
    __syncthreads();

    // ---------- phase 7: S^T += (k^T @ uhat)^T (all waves, 4 tiles each) ----------
#pragma unroll
    for (int t4 = 0; t4 < 4; ++t4) {
      const int tile = wv * 4 + t4;        // 0..31 over 16x2 tile grid
      const int r0 = (tile >> 1) * 16, n0 = (tile & 1) * 16;
      v16h a = frag_mk16(&sKT[0][0], CHUNK_, r0, 0, lane);  // A[m][kk] = k[kk][m]
      v16h b = frag_mk16(&sUhT[0][0], CHUNK_, n0, 0, lane);
      v8f acc = {};
      acc = wmma16(a, b, acc);
      const int n = n0 + (lane & 15), hi = (lane >> 4) & 1;
      float* sp = &sST[n][r0 + 8 * hi];
      v8f cur = *(v8f*)sp;
      cur += acc;
      *(v8f*)sp = cur;
    }
    __syncthreads();
  }
}

// ---------------------------------------------------------------------------
// Dual EMA scan: thread per (b,d), sequential over L.
// ---------------------------------------------------------------------------
__global__ void ema_kernel(const float* __restrict__ v, const float* __restrict__ gs,
                           const float* __restrict__ gl, float* __restrict__ es,
                           float* __restrict__ el) {
  int t = blockIdx.x * blockDim.x + threadIdx.x;
  if (t >= B_ * D_) return;
  const int b = t / D_, d = t % D_, h = d / DV_;
  float s0 = 0.f, s1 = 0.f;
  for (int l = 0; l < L_; ++l) {
    size_t ix = ((size_t)b * L_ + l) * D_ + d;
    float vv = v[ix];
    float g0 = gs[((size_t)b * L_ + l) * H_ + h];
    float g1 = gl[((size_t)b * L_ + l) * H_ + h];
    s0 = g0 * s0 + (1.f - g0) * vv;
    s1 = g1 * s1 + (1.f - g1) * vv;
    es[ix] = s0;
    el[ix] = s1;
  }
}

// ---------------------------------------------------------------------------
// Gate mix + per-head RMSNorm. One wave per (b,l,h).
// ---------------------------------------------------------------------------
__global__ __launch_bounds__(256) void combine_kernel(
    const float* __restrict__ v, const float* __restrict__ es,
    const float* __restrict__ dd, const float* __restrict__ el,
    const float* __restrict__ wts, const float* __restrict__ onw,
    float* __restrict__ O) {
  const int row = blockIdx.x * 8 + (threadIdx.x >> 5);  // over B*L*H
  const int lane = threadIdx.x & 31;
  const int h = row & (H_ - 1);
  const int bl = row >> 2;
  const size_t base = (size_t)bl * D_ + (size_t)h * DV_;
  const float* wp = wts + (size_t)row * 4;
  const float w0 = wp[0], w1 = wp[1], w2 = wp[2], w3 = wp[3];
  float o[8], ss = 0.f;
#pragma unroll
  for (int i = 0; i < 8; ++i) {
    size_t ix = base + lane + 32 * i;
    float val = w0 * v[ix] + w1 * es[ix] + w2 * dd[ix] + w3 * el[ix];
    o[i] = val;
    ss += val * val;
  }
  for (int off = 16; off; off >>= 1) ss += __shfl_xor(ss, off, 32);
  const float r = rsqrtf(ss / (float)DV_ + 1e-5f);
#pragma unroll
  for (int i = 0; i < 8; ++i)
    O[base + lane + 32 * i] = o[i] * r * onw[lane + 32 * i];
}

// ---------------------------------------------------------------------------
extern "C" void kernel_launch(void* const* d_in, const int* in_sizes, int n_in,
                              void* d_out, int out_size, void* d_ws, size_t ws_size,
                              hipStream_t stream) {
  const float* x       = (const float*)d_in[0];
  const float* Wq      = (const float*)d_in[1];
  const float* Wk      = (const float*)d_in[2];
  const float* Wv      = (const float*)d_in[3];
  const float* cq      = (const float*)d_in[4];
  const float* ck      = (const float*)d_in[5];
  const float* cv      = (const float*)d_in[6];
  const float* Wb      = (const float*)d_in[7];
  const float* Wds     = (const float*)d_in[8];
  const float* bds     = (const float*)d_in[9];
  const float* Wdl     = (const float*)d_in[10];
  const float* bdl     = (const float*)d_in[11];
  const float* Wtrunk  = (const float*)d_in[12];
  const float* btrunk  = (const float*)d_in[13];
  const float* Wcoarse = (const float*)d_in[14];
  const float* bcoarse = (const float*)d_in[15];
  const float* Wlocal  = (const float*)d_in[16];
  const float* blocal  = (const float*)d_in[17];
  const float* Wglobal = (const float*)d_in[18];
  const float* bglobal = (const float*)d_in[19];
  const float* ltc     = (const float*)d_in[20];
  const float* ltf     = (const float*)d_in[21];
  const float* onw     = (const float*)d_in[22];
  const float* Wo      = (const float*)d_in[23];

  const size_t BLD = (size_t)B_ * L_ * D_;
  float* p = (float*)d_ws;
  float* buf0 = p; p += BLD;                 // q_pre  -> delta_o
  float* buf1 = p; p += BLD;                 // k_pre  -> ema_s
  float* buf2 = p; p += BLD;                 // v_pre  -> ema_l
  float* buf3 = p; p += BLD;                 // q_conv -> O (pre-Wo)
  float* buf4 = p; p += BLD;                 // k_conv
  float* buf5 = p; p += BLD;                 // v_conv
  float* zbuf = p; p += (size_t)B_ * L_ * HID_;
  float* beta = p; p += (size_t)B_ * L_ * H_;
  float* gsb  = p; p += (size_t)B_ * L_ * H_;
  float* glb  = p; p += (size_t)B_ * L_ * H_;
  float* wts  = p; p += (size_t)B_ * L_ * H_ * 4;

  const dim3 blk(256);
  const dim3 gD(D_ / 128, (B_ * L_) / 128);
  const dim3 gH(HID_ / 128, (B_ * L_) / 128);
  const int totalBLD = B_ * L_ * D_;

  gemm_xwt<0><<<gD, blk, 0, stream>>>(x, Wq, nullptr, buf0, B_ * L_, D_, D_);
  gemm_xwt<0><<<gD, blk, 0, stream>>>(x, Wk, nullptr, buf1, B_ * L_, D_, D_);
  gemm_xwt<0><<<gD, blk, 0, stream>>>(x, Wv, nullptr, buf2, B_ * L_, D_, D_);
  gemm_xwt<1><<<gH, blk, 0, stream>>>(x, Wtrunk, btrunk, zbuf, B_ * L_, HID_, D_);

  conv_silu_kernel<<<4096, blk, 0, stream>>>(buf0, cq, buf3, totalBLD);
  conv_silu_kernel<<<4096, blk, 0, stream>>>(buf1, ck, buf4, totalBLD);
  conv_silu_kernel<<<4096, blk, 0, stream>>>(buf2, cv, buf5, totalBLD);

  smallproj_kernel<<<(B_ * L_) / 8, blk, 0, stream>>>(x, Wb, Wds, bds, Wdl, bdl,
                                                      beta, gsb, glb);
  gate_kernel<<<(B_ * L_) / 8, blk, 0, stream>>>(zbuf, Wcoarse, bcoarse, Wlocal,
                                                 blocal, Wglobal, bglobal, ltc, ltf,
                                                 wts);

  delta_kernel<<<B_ * H_ * SPLIT_, blk, 0, stream>>>(buf3, buf4, buf5, beta, buf0);

  ema_kernel<<<(B_ * D_) / 256, blk, 0, stream>>>(buf5, gsb, glb, buf1, buf2);

  combine_kernel<<<(B_ * L_ * H_) / 8, blk, 0, stream>>>(buf5, buf1, buf0, buf2,
                                                         wts, onw, buf3);

  gemm_xwt<0><<<gD, blk, 0, stream>>>(buf3, Wo, nullptr, (float*)d_out, B_ * L_,
                                      D_, D_);
}
